// MultiHeadAttention_84885733638276
// MI455X (gfx1250) — matrix operand order repaired and checked by reading, not compiled, not run on hardware
//
#include <hip/hip_runtime.h>
#include <hip/hip_bf16.h>
#include <math.h>

#define DEV __device__ __forceinline__

typedef __attribute__((ext_vector_type(16))) __bf16 v16bf;
typedef __attribute__((ext_vector_type(8)))  __bf16 v8bf;
typedef __attribute__((ext_vector_type(8)))  float  v8f;

namespace cfg {
constexpr int B = 4, S = 2048, D = 1024, H = 16, HD = 64;
constexpr float SCALE = 0.03125f; // 1/sqrt(D) = 1/sqrt(1024), per reference
}

union BF16x16 { v16bf v; v8bf h[2]; };

DEV v8f wmma_bf16(v16bf a, v16bf b, v8f c) {
  // v_wmma_f32_16x16x32_bf16: D = A(16x32) * B(32x16) + C, f32 accumulate
  return __builtin_amdgcn_wmma_f32_16x16x32_bf16(false, a, false, b, (short)0, c,
                                                 false, false);
}

// A-fragment (16x32 bf16): lane m = lane&15, g = lane>>4;
// 16 contiguous halves at K offset g*16 (matches B-fragment order).
DEV v16bf load_afrag_bf16(const __bf16* row, int g) {
  BF16x16 r;
  r.h[0] = *(const v8bf*)(row + g * 16);
  r.h[1] = *(const v8bf*)(row + g * 16 + 8);
  return r.v;
}
DEV v16bf load_afrag_f32(const float* row, int g) {
  v16bf a;
#pragma unroll
  for (int i = 0; i < 8; ++i) {
    a[i]     = (__bf16)row[g * 16 + i];
    a[8 + i] = (__bf16)row[g * 16 + 8 + i];
  }
  return a;
}
// B-fragment (32x16 bf16): lane n = lane&15, g = lane>>4; element e = 2p+j has
// K = g*16 + e => 16 contiguous halves per lane when stored K-contiguous.

// ---------------------------------------------------------------------------
// GEMM: Out[M,N] = A[M,K] (fp32 or bf16) * W[K,N] (fp32, cvt->bf16) [+ bias]
// 128 threads (4 waves), block tile 64(M) x 64(N), K-step 64 (8 WMMA / stage).
// ---------------------------------------------------------------------------
template <bool A_BF16, bool OUT_F32, bool BIAS>
__global__ __launch_bounds__(128)
void gemm_wmma_kernel(const void* __restrict__ Ap, const float* __restrict__ W,
                      const float* __restrict__ bias, void* __restrict__ Outp,
                      int M, int N, int K) {
  __shared__ __align__(64) __bf16 Wt[64 * 64];  // W tile transposed: [n][k], 8 KB
  const int tid  = threadIdx.x;
  const int wid  = tid >> 5;
  const int lane = tid & 31;
  const int g    = lane >> 4;
  const int lr   = lane & 15;
  const int m0   = blockIdx.x * 64 + wid * 16;
  const int n0   = blockIdx.y * 64;

  v8f acc[4];
#pragma unroll
  for (int nt = 0; nt < 4; ++nt)
#pragma unroll
    for (int p = 0; p < 8; ++p) acc[nt][p] = 0.f;

  for (int k0 = 0; k0 < K; k0 += 64) {
    {  // cooperative stage of W[k0:k0+64, n0:n0+64] into LDS as bf16 [n][k]
      const int r  = tid >> 4;        // 0..7
      const int c0 = (tid & 15) * 4;  // 0..60
#pragma unroll
      for (int it = 0; it < 8; ++it) {
        const int rr = r + it * 8;  // 0..63
        const float4 wv = *(const float4*)(W + (size_t)(k0 + rr) * N + n0 + c0);
        Wt[(c0 + 0) * 64 + rr] = (__bf16)wv.x;
        Wt[(c0 + 1) * 64 + rr] = (__bf16)wv.y;
        Wt[(c0 + 2) * 64 + rr] = (__bf16)wv.z;
        Wt[(c0 + 3) * 64 + rr] = (__bf16)wv.w;
      }
    }
    if (k0 + 64 < K)  // hint next W tile into cache (global_prefetch_b8)
      __builtin_prefetch(W + (size_t)(k0 + 64 + (tid >> 4)) * N + n0 + (tid & 15) * 4, 0, 1);
    __syncthreads();

    v16bf a0, a1;
    if (A_BF16) {
      const __bf16* ar = (const __bf16*)Ap + (size_t)(m0 + lr) * K + k0;
      a0 = load_afrag_bf16(ar, g);
      a1 = load_afrag_bf16(ar + 32, g);
    } else {
      const float* ar = (const float*)Ap + (size_t)(m0 + lr) * K + k0;
      a0 = load_afrag_f32(ar, g);
      a1 = load_afrag_f32(ar + 32, g);
    }

#pragma unroll
    for (int nt = 0; nt < 4; ++nt) {
      const __bf16* wc = &Wt[(nt * 16 + lr) * 64];
      acc[nt] = wmma_bf16(a0, *(const v16bf*)(wc + g * 16), acc[nt]);
      acc[nt] = wmma_bf16(a1, *(const v16bf*)(wc + 32 + g * 16), acc[nt]);
    }
    __syncthreads();
  }

  // C/D layout: VGPR p -> row m = g*8 + p, lane lr -> col
#pragma unroll
  for (int nt = 0; nt < 4; ++nt) {
#pragma unroll
    for (int p = 0; p < 8; ++p) {
      const size_t row = (size_t)m0 + g * 8 + p;
      const size_t col = (size_t)n0 + nt * 16 + lr;
      float v = acc[nt][p];
      if (BIAS) v += bias[col];
      if (OUT_F32)
        ((float*)Outp)[row * N + col] = v;
      else
        ((__bf16*)Outp)[row * N + col] = (__bf16)v;
    }
  }
}

// ---------------------------------------------------------------------------
// Flash attention: one (batch, head, 64-query block) per 128-thread workgroup.
// Each wave owns 16 query rows; online softmax over S keys in chunks of 64.
// K and V chunks staged once per workgroup in LDS (shared by all 4 waves);
// K needs no transpose (B-frag reduction dim is K's row-contiguous dim),
// V is transposed to [d][key] during staging.
// ---------------------------------------------------------------------------
__global__ __launch_bounds__(128)
void flash_attn_kernel(const void* __restrict__ Qp, const void* __restrict__ Kp,
                       const void* __restrict__ Vp, void* __restrict__ Op) {
  using namespace cfg;
  const __bf16* Q = (const __bf16*)Qp;
  const __bf16* K = (const __bf16*)Kp;
  const __bf16* V = (const __bf16*)Vp;

  __shared__ __align__(64) __bf16 Kt[64 * 64];      // K chunk [key][d], 8 KB
  __shared__ __align__(64) __bf16 Vt[64 * 64];      // V chunk transposed [d][key], 8 KB
  __shared__ __align__(64) __bf16 Pb[4 * 16 * 64];  // per-wave P tile [m][key], 8 KB

  const int tid  = threadIdx.x;
  const int wid  = tid >> 5;
  const int lane = tid & 31;
  const int g    = lane >> 4;
  const int lr   = lane & 15;
  const int h    = blockIdx.y;
  const size_t bs = (size_t)blockIdx.z * S;
  const int q0   = blockIdx.x * 64 + wid * 16;
  const int hoff = h * HD;

  // Q A-fragments for this wave's 16 rows (HD=64 -> two reduction steps of 32)
  const __bf16* qrow = Q + (bs + q0 + lr) * D + hoff;
  const v16bf aq0 = load_afrag_bf16(qrow, g);
  const v16bf aq1 = load_afrag_bf16(qrow + 32, g);

  float rmax[8], rsum[8], alpha[8];
  v8f o[4];
#pragma unroll
  for (int p = 0; p < 8; ++p) { rmax[p] = -__builtin_inff(); rsum[p] = 0.f; }
#pragma unroll
  for (int dt = 0; dt < 4; ++dt)
#pragma unroll
    for (int p = 0; p < 8; ++p) o[dt][p] = 0.f;

  for (int j0 = 0; j0 < S; j0 += 64) {
    __syncthreads();
    {  // stage K (copy) and V (transpose) chunks into LDS
      const int kk = tid >> 1;        // key row 0..63
      const int d0 = (tid & 1) * 32;  // d column group
      const size_t rowbase = (bs + j0 + kk) * D + hoff + d0;

      const v16bf k0v = *(const v16bf*)(K + rowbase);
      const v16bf k1v = *(const v16bf*)(K + rowbase + 16);
      *(v16bf*)(&Kt[kk * 64 + d0])      = k0v;
      *(v16bf*)(&Kt[kk * 64 + d0 + 16]) = k1v;

      const v16bf v0 = *(const v16bf*)(V + rowbase);
      const v16bf v1 = *(const v16bf*)(V + rowbase + 16);
#pragma unroll
      for (int i = 0; i < 16; ++i) {
        Vt[(d0 + i) * 64 + kk]      = v0[i];
        Vt[(d0 + 16 + i) * 64 + kk] = v1[i];
      }
      if (j0 + 64 < S) {  // hint next K/V chunk into cache
        __builtin_prefetch(K + rowbase + (size_t)64 * D, 0, 1);
        __builtin_prefetch(V + rowbase + (size_t)64 * D, 0, 1);
      }
    }
    __syncthreads();

    // scores S = Q * K^T for four 16-key tiles; B-frags from LDS, shared by waves
    v8f s[4];
#pragma unroll
    for (int t = 0; t < 4; ++t) {
      const __bf16* krow = &Kt[(t * 16 + lr) * 64];
      const v16bf b0 = *(const v16bf*)(krow + g * 16);
      const v16bf b1 = *(const v16bf*)(krow + 32 + g * 16);
      v8f c;
#pragma unroll
      for (int p = 0; p < 8; ++p) c[p] = 0.f;
      c = wmma_bf16(aq0, b0, c);
      c = wmma_bf16(aq1, b1, c);
      s[t] = c;
    }

    // online softmax; row m = g*8 + p lives in one 16-lane half
    __bf16* pbw = &Pb[wid * (16 * 64)];
#pragma unroll
    for (int p = 0; p < 8; ++p) {
      float sv[4];
#pragma unroll
      for (int t = 0; t < 4; ++t) sv[t] = s[t][p] * SCALE;
      float mx = fmaxf(fmaxf(sv[0], sv[1]), fmaxf(sv[2], sv[3]));
      mx = fmaxf(mx, __shfl_xor(mx, 1));
      mx = fmaxf(mx, __shfl_xor(mx, 2));
      mx = fmaxf(mx, __shfl_xor(mx, 4));
      mx = fmaxf(mx, __shfl_xor(mx, 8));
      const float mnew = fmaxf(rmax[p], mx);
      alpha[p] = __expf(rmax[p] - mnew);
      rmax[p]  = mnew;
      float e[4], rs = 0.f;
#pragma unroll
      for (int t = 0; t < 4; ++t) { e[t] = __expf(sv[t] - mnew); rs += e[t]; }
      rs += __shfl_xor(rs, 1);
      rs += __shfl_xor(rs, 2);
      rs += __shfl_xor(rs, 4);
      rs += __shfl_xor(rs, 8);
      rsum[p] = rsum[p] * alpha[p] + rs;
#pragma unroll
      for (int t = 0; t < 4; ++t)  // C-layout -> LDS [m][key]
        pbw[(g * 8 + p) * 64 + t * 16 + lr] = (__bf16)e[t];
    }
    __syncthreads();  // P tiles visible (and Kt reads done) before next phase

    // rescale running output, then O += P(16x64) @ Vchunk(64x64)
#pragma unroll
    for (int dt = 0; dt < 4; ++dt)
#pragma unroll
      for (int p = 0; p < 8; ++p) o[dt][p] *= alpha[p];

    const v16bf ap0 = load_afrag_bf16(pbw + lr * 64, g);
    const v16bf ap1 = load_afrag_bf16(pbw + lr * 64 + 32, g);
#pragma unroll
    for (int dt = 0; dt < 4; ++dt) {
      const __bf16* vcol = &Vt[(dt * 16 + lr) * 64];
      o[dt] = wmma_bf16(ap0, *(const v16bf*)(vcol + g * 16), o[dt]);
      o[dt] = wmma_bf16(ap1, *(const v16bf*)(vcol + 32 + g * 16), o[dt]);
    }
  }

  // normalize and store attention output (bf16) as [B,S,D]
  __bf16* O = (__bf16*)Op;
#pragma unroll
  for (int p = 0; p < 8; ++p) {
    const float inv = 1.f / rsum[p];
    const size_t row = bs + q0 + g * 8 + p;
#pragma unroll
    for (int dt = 0; dt < 4; ++dt)
      O[row * D + hoff + dt * 16 + lr] = (__bf16)(o[dt][p] * inv);
  }
}

// ---------------------------------------------------------------------------
extern "C" void kernel_launch(void* const* d_in, const int* in_sizes, int n_in,
                              void* d_out, int out_size, void* d_ws, size_t ws_size,
                              hipStream_t stream) {
  using namespace cfg;
  const float* q  = (const float*)d_in[0];
  const float* k  = (const float*)d_in[1];
  const float* v  = (const float*)d_in[2];
  const float* Wq = (const float*)d_in[3];
  const float* Wk = (const float*)d_in[4];
  const float* Wv = (const float*)d_in[5];
  const float* Wo = (const float*)d_in[6];
  const float* bo = (const float*)d_in[7];

  char* ws = (char*)d_ws;
  const size_t eb = (size_t)B * S * D * 2;  // bf16 buffer bytes (16 MB each)
  void* Qb = ws;
  void* Kb = ws + eb;
  void* Vb = ws + 2 * eb;
  void* Ao = ws + 3 * eb;

  const dim3 th(128);
  const dim3 gproj(B * S / 64, D / 64);  // 128 x 16

  gemm_wmma_kernel<false, false, false><<<gproj, th, 0, stream>>>(q, Wq, nullptr, Qb, B * S, D, D);
  gemm_wmma_kernel<false, false, false><<<gproj, th, 0, stream>>>(k, Wk, nullptr, Kb, B * S, D, D);
  gemm_wmma_kernel<false, false, false><<<gproj, th, 0, stream>>>(v, Wv, nullptr, Vb, B * S, D, D);

  flash_attn_kernel<<<dim3(S / 64, H, B), th, 0, stream>>>(Qb, Kb, Vb, Ao);

  gemm_wmma_kernel<true, true, true><<<gproj, th, 0, stream>>>(Ao, Wo, bo, d_out, B * S, D, D);
}
